// EnergyTransformer_57672820851129
// MI455X (gfx1250) — matrix operand order repaired and checked by reading, not compiled, not run on hardware
//
#include <hip/hip_runtime.h>
#include <hip/hip_bf16.h>

typedef __attribute__((ext_vector_type(16))) __bf16 v16bf;
typedef __attribute__((ext_vector_type(8)))  float  v8f;
typedef __attribute__((ext_vector_type(4)))  unsigned int u32x4;

#define ET_ALPHA 0.1f
#define ET_LN_EPS 1e-5f
#define ET_STEPS 12

// ---------- bf16 helpers (raw u16, no scalar __bf16 math) ----------
__device__ inline float bf2f(unsigned short u) {
  unsigned int x = ((unsigned int)u) << 16;
  return __builtin_bit_cast(float, x);
}
__device__ inline unsigned short f2bf(float f) {
  unsigned int x = __builtin_bit_cast(unsigned int, f);
  unsigned int lsb = (x >> 16) & 1u;
  x += 0x7fffu + lsb; // round to nearest even
  return (unsigned short)(x >> 16);
}

union Frag16 {
  v16bf v;
  unsigned int w[8];
  unsigned short u[16];
  u32x4 q[2];
};

// ---------- batched WMMA GEMM ----------
// C[m,n] = scale * sum_k Aelem(m,k)*Belem(k,n)
//   AM=0: A row-major [M,K]  (K contiguous -> b128 fast path)
//   AM=1: A stored [K,M]     (strided element loads)
//   BM=0: B stored [N,K]     (K contiguous -> b128 fast path; "NT")
//   BM=1: B stored [K,N]     (strided element loads; "NN")
// EPI: 0 = store bf16, 1 = relu->bf16, 2 = f32 +=, 3 = store f32
// batch z = (b,h) = (z/batchH, z%batchH) with two-level pointer strides.
// Requirements (met by all launches): K%32==0, M%16==0, N%16==0,
// all row strides are multiples of 8 elements (16B alignment).
template<int AM, int BM, int EPI>
__global__ __launch_bounds__(256) void gemm_wmma(
    const unsigned short* __restrict__ A, const unsigned short* __restrict__ B,
    void* __restrict__ C,
    int M, int N, int K, int lda, int ldb, int ldc, int batchH,
    long sAb, long sAh, long sBb, long sBh, long sCb, long sCh, float scale) {
  int z = blockIdx.z;
  int bb = z / batchH, hh = z - bb * batchH;
  A += bb * sAb + hh * sAh;
  B += bb * sBb + hh * sBh;
  long coff = bb * sCb + hh * sCh;

  int tid = threadIdx.x;
  int lane = tid & 31;
  int wave = tid >> 5;
  int wr = wave >> 1;  // 0..3
  int wc = wave & 1;   // 0..1
  long rowBase = (long)blockIdx.x * 128 + wr * 32;
  long colBase = (long)blockIdx.y * 128 + wc * 64;

  int mlane = lane & 15;
  int khalfA = (lane >> 4) << 3;   // A: lanes 16-31 shift K by 8
  int khB = (lane >> 4) << 3;      // B: lanes 16-31 shift K by 8

  // Precomputed, clamped per-lane base pointers (no predication; clamped
  // rows/cols read valid duplicate data whose C entries are never stored).
  const unsigned short* aPtr[2];
#pragma unroll
  for (int i = 0; i < 2; ++i) {
    long rr = rowBase + i * 16 + mlane;
    rr = rr < (long)M ? rr : (long)M - 1;
    aPtr[i] = (AM == 0) ? (A + rr * (long)lda + khalfA) : (A + rr);
  }
  const unsigned short* bPtr[4];
#pragma unroll
  for (int j = 0; j < 4; ++j) {
    long cc = colBase + j * 16 + mlane;
    cc = cc < (long)N ? cc : (long)N - 1;
    bPtr[j] = (BM == 0) ? (B + cc * (long)ldb + khB) : (B + cc);
  }

  v8f acc[2][4];
  v8f zero8 = {0.f, 0.f, 0.f, 0.f, 0.f, 0.f, 0.f, 0.f};
#pragma unroll
  for (int i = 0; i < 2; ++i)
#pragma unroll
    for (int j = 0; j < 4; ++j) acc[i][j] = zero8;

  for (int k0 = 0; k0 < K; k0 += 32) {
    v16bf a[2], b[4];
#pragma unroll
    for (int i = 0; i < 2; ++i) {
      Frag16 f;
      if (AM == 0) {
        f.q[0] = *(const u32x4*)(aPtr[i] + k0);       // K = khalf + 0..7
        f.q[1] = *(const u32x4*)(aPtr[i] + k0 + 16);  // K = 16 + khalf + 0..7
      } else {
#pragma unroll
        for (int p = 0; p < 8; ++p) {
          int kk = k0 + ((p >> 2) << 4) + khalfA + ((p & 3) << 1);
          f.u[2 * p + 0] = aPtr[i][(long)(kk + 0) * lda];
          f.u[2 * p + 1] = aPtr[i][(long)(kk + 1) * lda];
        }
      }
      a[i] = f.v;
    }
#pragma unroll
    for (int j = 0; j < 4; ++j) {
      Frag16 f;
      if (BM == 0) {
        f.q[0] = *(const u32x4*)(bPtr[j] + k0);      // K = kh + 0..7
        f.q[1] = *(const u32x4*)(bPtr[j] + k0 + 16); // K = 16 + kh + 0..7
      } else {
#pragma unroll
        for (int p = 0; p < 8; ++p) {
          int kk = k0 + ((p >> 2) << 4) + khB + ((p & 3) << 1);
          f.u[2 * p + 0] = bPtr[j][(long)(kk + 0) * ldb];
          f.u[2 * p + 1] = bPtr[j][(long)(kk + 1) * ldb];
        }
      }
      b[j] = f.v;
    }

    acc[0][0] = __builtin_amdgcn_wmma_f32_16x16x32_bf16(false, a[0], false, b[0], (short)0, acc[0][0], false, false);
    acc[0][1] = __builtin_amdgcn_wmma_f32_16x16x32_bf16(false, a[0], false, b[1], (short)0, acc[0][1], false, false);
    acc[0][2] = __builtin_amdgcn_wmma_f32_16x16x32_bf16(false, a[0], false, b[2], (short)0, acc[0][2], false, false);
    acc[0][3] = __builtin_amdgcn_wmma_f32_16x16x32_bf16(false, a[0], false, b[3], (short)0, acc[0][3], false, false);
    acc[1][0] = __builtin_amdgcn_wmma_f32_16x16x32_bf16(false, a[1], false, b[0], (short)0, acc[1][0], false, false);
    acc[1][1] = __builtin_amdgcn_wmma_f32_16x16x32_bf16(false, a[1], false, b[1], (short)0, acc[1][1], false, false);
    acc[1][2] = __builtin_amdgcn_wmma_f32_16x16x32_bf16(false, a[1], false, b[2], (short)0, acc[1][2], false, false);
    acc[1][3] = __builtin_amdgcn_wmma_f32_16x16x32_bf16(false, a[1], false, b[3], (short)0, acc[1][3], false, false);
  }

  // C layout: VGPR e -> m = e + (lane>>4)*8 ; n = lane&15
  int cl = lane & 15;
  int rh = (lane >> 4) << 3;
#pragma unroll
  for (int i = 0; i < 2; ++i) {
#pragma unroll
    for (int j = 0; j < 4; ++j) {
      // uniform (scalar) guards: M,N are multiples of 16 in all launches
      if ((rowBase + i * 16) < (long)M && (colBase + j * 16) < (long)N) {
        long r0 = rowBase + i * 16 + rh;
        long c0 = colBase + j * 16 + cl;
#pragma unroll
        for (int e = 0; e < 8; ++e) {
          float val = acc[i][j][e] * scale;
          long idx = coff + (r0 + e) * (long)ldc + c0;
          if (EPI == 0) {
            ((unsigned short*)C)[idx] = f2bf(val);
          } else if (EPI == 1) {
            ((unsigned short*)C)[idx] = f2bf(val > 0.f ? val : 0.f);
          } else if (EPI == 2) {
            ((float*)C)[idx] += val;
          } else {
            ((float*)C)[idx] = val;
          }
        }
      }
    }
  }
}

// ---------- LayerNorm: one block per row, writes bf16 g ----------
__global__ __launch_bounds__(256) void ln_kernel(
    const float* __restrict__ x, const float* __restrict__ gamma,
    const float* __restrict__ delta, unsigned short* __restrict__ g, int D) {
  __shared__ float red[256];
  long row = blockIdx.x;
  const float* xr = x + row * D;
  int tid = threadIdx.x;

  float s = 0.f;
  for (int d = tid; d < D; d += 256) s += xr[d];
  red[tid] = s;
  __syncthreads();
  for (int off = 128; off > 0; off >>= 1) {
    if (tid < off) red[tid] += red[tid + off];
    __syncthreads();
  }
  float mu = red[0] / (float)D;
  __syncthreads();

  float v = 0.f;
  for (int d = tid; d < D; d += 256) {
    float t = xr[d] - mu;
    v += t * t;
  }
  red[tid] = v;
  __syncthreads();
  for (int off = 128; off > 0; off >>= 1) {
    if (tid < off) red[tid] += red[tid + off];
    __syncthreads();
  }
  float rstd = rsqrtf(red[0] / (float)D + ET_LN_EPS);

  unsigned short* gr = g + row * D;
  for (int d = tid; d < D; d += 256)
    gr[d] = f2bf(gamma[d] * (xr[d] - mu) * rstd + delta[d]);
}

// ---------- row softmax over bf16, in place ----------
__global__ __launch_bounds__(256) void softmax_kernel(unsigned short* __restrict__ S, int L) {
  __shared__ float red[256];
  long row = blockIdx.x;
  unsigned short* sr = S + row * (long)L;
  int tid = threadIdx.x;

  float m = -3.4e38f;
  for (int i = tid; i < L; i += 256) m = fmaxf(m, bf2f(sr[i]));
  red[tid] = m;
  __syncthreads();
  for (int off = 128; off > 0; off >>= 1) {
    if (tid < off) red[tid] = fmaxf(red[tid], red[tid + off]);
    __syncthreads();
  }
  float mx = red[0];
  __syncthreads();

  float s = 0.f;
  for (int i = tid; i < L; i += 256) s += __expf(bf2f(sr[i]) - mx);
  red[tid] = s;
  __syncthreads();
  for (int off = 128; off > 0; off >>= 1) {
    if (tid < off) red[tid] += red[tid + off];
    __syncthreads();
  }
  float inv = 1.f / red[0];

  for (int i = tid; i < L; i += 256)
    sr[i] = f2bf(__expf(bf2f(sr[i]) - mx) * inv);
}

// ---------- misc elementwise ----------
__global__ void f32_to_bf16_kernel(const float* __restrict__ in,
                                   unsigned short* __restrict__ out, long n) {
  long i = (long)blockIdx.x * 256 + threadIdx.x;
  if (i < n) out[i] = f2bf(in[i]);
}

// in[R][C] (f32) -> out[C][R] (bf16)
__global__ void transpose_f32_to_bf16_kernel(const float* __restrict__ in,
                                             unsigned short* __restrict__ out,
                                             int R, int Ccols) {
  long i = (long)blockIdx.x * 256 + threadIdx.x;
  long total = (long)R * Ccols;
  if (i < total) {
    long r = i / Ccols, c = i - r * Ccols;
    out[c * (long)R + r] = f2bf(in[i]);
  }
}

__global__ void update_kernel(float* __restrict__ x, const float* __restrict__ grad,
                              long n, float alpha) {
  long i = (long)blockIdx.x * 256 + threadIdx.x;
  if (i < n) x[i] += alpha * grad[i];
}

// ---------- host ----------
extern "C" void kernel_launch(void* const* d_in, const int* in_sizes, int n_in,
                              void* d_out, int out_size, void* d_ws, size_t ws_size,
                              hipStream_t stream) {
  const int B = 8, N = 512, D = 768, H = 12, Y = 64, M = 3072;
  const int BN = B * N;       // 4096
  const int HY = H * Y;       // 768
  const float beta = 0.125f;  // 1/sqrt(64)

  const float* x_in  = (const float*)d_in[0];
  const float* Wq    = (const float*)d_in[1];
  const float* Wk    = (const float*)d_in[2];
  const float* xi    = (const float*)d_in[3];
  const float* gamma = (const float*)d_in[4];
  const float* delta = (const float*)d_in[5];
  float* x = (float*)d_out;

  char* ws = (char*)d_ws;
  size_t off = 0;
  auto alloc = [&](size_t bytes) -> char* {
    char* p = ws + off;
    off = (off + bytes + 255) & ~(size_t)255;
    return p;
  };

  unsigned short* Gbf   = (unsigned short*)alloc((size_t)BN * D * 2);
  unsigned short* Wqbf  = (unsigned short*)alloc((size_t)HY * D * 2);
  unsigned short* Wkbf  = (unsigned short*)alloc((size_t)HY * D * 2);
  unsigned short* xibf  = (unsigned short*)alloc((size_t)M * D * 2);
  unsigned short* WqTbf = (unsigned short*)alloc((size_t)D * HY * 2);  // [d][hy]
  unsigned short* WkTbf = (unsigned short*)alloc((size_t)D * HY * 2);  // [d][hy]
  unsigned short* xiTbf = (unsigned short*)alloc((size_t)D * M * 2);   // [d][m]
  unsigned short* Qbf   = (unsigned short*)alloc((size_t)BN * HY * 2);
  unsigned short* Kbf   = (unsigned short*)alloc((size_t)BN * HY * 2);
  unsigned short* Sbf   = (unsigned short*)alloc((size_t)B * H * N * N * 2);  // 50 MB
  unsigned short* hidbf = Sbf;  // reused after dq/dk consume P (25 MB <= 50 MB)
  unsigned short* dqbf  = (unsigned short*)alloc((size_t)BN * HY * 2);
  unsigned short* dkbf  = (unsigned short*)alloc((size_t)BN * HY * 2);
  float* grad = (float*)alloc((size_t)BN * D * 4);

  // x lives in d_out across steps
  hipMemcpyAsync(x, x_in, (size_t)BN * D * sizeof(float), hipMemcpyDeviceToDevice, stream);

  // static weights: bf16 copies + transposed bf16 copies (once per call)
  {
    long n1 = (long)HY * D, n3 = (long)M * D;
    unsigned g1 = (unsigned)((n1 + 255) / 256), g3 = (unsigned)((n3 + 255) / 256);
    f32_to_bf16_kernel<<<dim3(g1), 256, 0, stream>>>(Wq, Wqbf, n1);
    f32_to_bf16_kernel<<<dim3(g1), 256, 0, stream>>>(Wk, Wkbf, n1);
    f32_to_bf16_kernel<<<dim3(g3), 256, 0, stream>>>(xi, xibf, n3);
    transpose_f32_to_bf16_kernel<<<dim3(g1), 256, 0, stream>>>(Wq, WqTbf, HY, D);
    transpose_f32_to_bf16_kernel<<<dim3(g1), 256, 0, stream>>>(Wk, WkTbf, HY, D);
    transpose_f32_to_bf16_kernel<<<dim3(g3), 256, 0, stream>>>(xi, xiTbf, M, D);
  }

  for (int step = 0; step < ET_STEPS; ++step) {
    // g = LN(x) -> bf16
    ln_kernel<<<dim3(BN), 256, 0, stream>>>(x, gamma, delta, Gbf, D);

    // Q = G @ Wq^T ; K = G @ Wk^T  (fast NT), [4096 x 768]
    gemm_wmma<0, 0, 0><<<dim3(32, 6, 1), 256, 0, stream>>>(
        Gbf, Wqbf, Qbf, BN, HY, D, D, D, HY, 1, 0, 0, 0, 0, 0, 0, 1.0f);
    gemm_wmma<0, 0, 0><<<dim3(32, 6, 1), 256, 0, stream>>>(
        Gbf, Wkbf, Kbf, BN, HY, D, D, D, HY, 1, 0, 0, 0, 0, 0, 0, 1.0f);

    // S[b,h] = beta * Q_h @ K_h^T  (fast NT), batched z = b*H + h
    gemm_wmma<0, 0, 0><<<dim3(4, 4, B * H), 256, 0, stream>>>(
        Qbf, Kbf, Sbf, N, N, Y, HY, HY, N, H,
        (long)N * HY, (long)Y, (long)N * HY, (long)Y,
        (long)H * N * N, (long)N * N, beta);

    // P = softmax(S) row-wise, in place (bf16)
    softmax_kernel<<<dim3(B * H * N), 256, 0, stream>>>(Sbf, N);

    // dq_h = P_h @ K_h  (A fast, B strided), packed into [B,N,H*Y]
    gemm_wmma<0, 1, 0><<<dim3(4, 1, B * H), 256, 0, stream>>>(
        Sbf, Kbf, dqbf, N, Y, N, N, HY, HY, H,
        (long)H * N * N, (long)N * N, (long)N * HY, (long)Y,
        (long)N * HY, (long)Y, 1.0f);

    // dk_h = P_h^T @ Q_h  (A strided, B strided)
    gemm_wmma<1, 1, 0><<<dim3(4, 1, B * H), 256, 0, stream>>>(
        Sbf, Qbf, dkbf, N, Y, N, N, HY, HY, H,
        (long)H * N * N, (long)N * N, (long)N * HY, (long)Y,
        (long)N * HY, (long)Y, 1.0f);

    // hid = relu(G @ xi^T)  (fast NT), [4096 x 3072], reuses P buffer
    gemm_wmma<0, 0, 1><<<dim3(32, 24, 1), 256, 0, stream>>>(
        Gbf, xibf, hidbf, BN, M, D, D, D, M, 1, 0, 0, 0, 0, 0, 0, 1.0f);

    // grad  = dq @ WqFlat   (fast NT via WqT, store f32)
    gemm_wmma<0, 0, 3><<<dim3(32, 6, 1), 256, 0, stream>>>(
        dqbf, WqTbf, grad, BN, D, HY, HY, HY, D, 1, 0, 0, 0, 0, 0, 0, 1.0f);
    // grad += dk @ WkFlat   (fast NT via WkT)
    gemm_wmma<0, 0, 2><<<dim3(32, 6, 1), 256, 0, stream>>>(
        dkbf, WkTbf, grad, BN, D, HY, HY, HY, D, 1, 0, 0, 0, 0, 0, 0, 1.0f);
    // grad += hid @ xi      (fast NT via xiT)
    gemm_wmma<0, 0, 2><<<dim3(32, 6, 1), 256, 0, stream>>>(
        hidbf, xiTbf, grad, BN, D, M, M, M, D, 1, 0, 0, 0, 0, 0, 0, 1.0f);

    // x += ALPHA * grad   (grad_E = -grad; update is x - ALPHA*grad_E)
    long n = (long)BN * D;
    update_kernel<<<dim3((unsigned)((n + 255) / 256)), 256, 0, stream>>>(x, grad, n, ET_ALPHA);
  }

  (void)in_sizes; (void)n_in; (void)out_size; (void)ws_size;
}
